// HistogramMatching_18511309046427
// MI455X (gfx1250) — compile-verified, matrix operand order checked by reading, not run on hardware
//
#include <hip/hip_runtime.h>

typedef __attribute__((ext_vector_type(16))) _Float16 v16h;
typedef __attribute__((ext_vector_type(8)))  float    v8f;

#define NBINS     256
#define CH_PIX    65536      // 256*256 pixels per (n,c) channel
#define CHUNKS    16
#define CHUNK_PIX 4096       // CH_PIX / CHUNKS
#define HIST_THREADS 512     // 16 wave32 waves -> one 16-bin group per wave

// ---------------------------------------------------------------------------
// Kernel 1: soft histogram via WMMA reduction.
// hist[b] = sum_p exp(-0.5*((x_p - b/255)/sigma)^2)
// A = 16x32 all-ones (f16), B = 32x16 Gaussian weights (f16), C/D = f32 acc.
// With A==1, D[m][n] = sum_k B[k][n]; lane n and lane n+16 each supply 16
// distinct pixels for bin column n, so the reduction is exact regardless of
// the ISA's internal k-slot ordering.
// ---------------------------------------------------------------------------
__global__ __launch_bounds__(HIST_THREADS)
void hist_wmma_kernel(const float* __restrict__ src,
                      const float* __restrict__ tgt,
                      float* __restrict__ partial)   // [12][CHUNKS][NBINS]
{
    __shared__ float s_pix[CHUNK_PIX];

    const int bid   = blockIdx.x;
    const int ic    = bid / CHUNKS;          // 0..11 image-channel (6 src + 6 tgt)
    const int chunk = bid % CHUNKS;
    const float* base = (ic < 6) ? (src + ic * CH_PIX)
                                 : (tgt + (ic - 6) * CH_PIX);
    base += chunk * CHUNK_PIX;

    // Stage this block's 4096-pixel chunk into LDS (float4 vectorized).
    const float4* g4 = (const float4*)base;
    float4*       s4 = (float4*)s_pix;
    for (int i = threadIdx.x; i < CHUNK_PIX / 4; i += HIST_THREADS)
        s4[i] = g4[i];
    __syncthreads();

    const int wave = threadIdx.x >> 5;       // bin group: bins [16w, 16w+16)
    const int lane = threadIdx.x & 31;
    const int n    = lane & 15;              // bin column within group
    const int half = lane >> 4;              // which 16 of the 32 pixels
    const float binv      = (float)(wave * 16 + n) * (1.0f / 255.0f);
    const float inv_sigma = 100.0f;                  // 1/0.01
    const float c2        = -0.72134752044f;         // -0.5/ln(2)

    v16h a;
    #pragma unroll
    for (int i = 0; i < 16; ++i) a[i] = (_Float16)1.0f;
    v8f acc = {0.f, 0.f, 0.f, 0.f, 0.f, 0.f, 0.f, 0.f};

    const float4* sp = (const float4*)s_pix;
    for (int it = 0; it < CHUNK_PIX / 32; ++it) {
        const int pixbase = it * 32 + half * 16;     // this lane's 16 pixels
        v16h b;
        #pragma unroll
        for (int q = 0; q < 4; ++q) {
            float4 p = sp[(pixbase >> 2) + q];
            float t0 = (p.x - binv) * inv_sigma;
            float t1 = (p.y - binv) * inv_sigma;
            float t2 = (p.z - binv) * inv_sigma;
            float t3 = (p.w - binv) * inv_sigma;
            b[4*q+0] = (_Float16)__builtin_amdgcn_exp2f(c2 * t0 * t0);
            b[4*q+1] = (_Float16)__builtin_amdgcn_exp2f(c2 * t1 * t1);
            b[4*q+2] = (_Float16)__builtin_amdgcn_exp2f(c2 * t2 * t2);
            b[4*q+3] = (_Float16)__builtin_amdgcn_exp2f(c2 * t3 * t3);
        }
        // D = A(ones) x B(weights) + C  -> per-bin partial sums in every row
        acc = __builtin_amdgcn_wmma_f32_16x16x32_f16(
                  false, a, false, b, (short)0, acc, false, false);
    }

    // Row M=0 of D lives in VGPR0 across lanes 0..15 (N = lane).
    if (lane < 16)
        partial[(ic * CHUNKS + chunk) * NBINS + wave * 16 + lane] = acc[0];
}

// ---------------------------------------------------------------------------
// Kernel 2: reduce chunk partials, normalize, inclusive cumsum -> CDFs.
// ---------------------------------------------------------------------------
__global__ __launch_bounds__(NBINS)
void cdf_kernel(const float* __restrict__ partial,  // [12][CHUNKS][NBINS]
                float* __restrict__ cdf_src,        // [6][NBINS] clipped
                float* __restrict__ cdf_tgt)        // [6][NBINS]
{
    __shared__ float red[NBINS];
    __shared__ float sh[NBINS];
    const int ch = blockIdx.x;    // 0..11
    const int b  = threadIdx.x;

    float h = 0.f;
    #pragma unroll
    for (int k = 0; k < CHUNKS; ++k)
        h += partial[(ch * CHUNKS + k) * NBINS + b];

    red[b] = h;
    __syncthreads();
    for (int off = NBINS / 2; off > 0; off >>= 1) {
        if (b < off) red[b] += red[b + off];
        __syncthreads();
    }
    const float norm = h / (red[0] + 1e-6f);

    sh[b] = norm;
    __syncthreads();
    for (int off = 1; off < NBINS; off <<= 1) {     // Hillis-Steele scan
        float t = (b >= off) ? sh[b - off] : 0.f;
        __syncthreads();
        sh[b] += t;
        __syncthreads();
    }
    float cdf = sh[b];
    if (ch < 6) cdf_src[ch * NBINS + b] = fminf(fmaxf(cdf, 0.f), 1.f);
    else        cdf_tgt[(ch - 6) * NBINS + b] = cdf;
}

// ---------------------------------------------------------------------------
// Kernel 3: per-pixel mapping: hard bin -> cdf_src value -> inverse target
// CDF via branchless upper_bound (== searchsorted side='right') + lerp.
// ---------------------------------------------------------------------------
__global__ __launch_bounds__(NBINS)
void map_kernel(const float* __restrict__ src,
                const float* __restrict__ cdf_src,
                const float* __restrict__ cdf_tgt,
                float* __restrict__ out)
{
    __shared__ float s_cs[NBINS];
    __shared__ float s_ct[NBINS];
    const int bid   = blockIdx.x;        // 6 channels * CHUNKS blocks
    const int ch    = bid / CHUNKS;
    const int chunk = bid % CHUNKS;
    s_cs[threadIdx.x] = cdf_src[ch * NBINS + threadIdx.x];
    s_ct[threadIdx.x] = cdf_tgt[ch * NBINS + threadIdx.x];
    __syncthreads();

    const int base = ch * CH_PIX + chunk * CHUNK_PIX;
    for (int i = threadIdx.x; i < CHUNK_PIX; i += NBINS) {
        float x = src[base + i];
        int bi = (int)(x * 255.0f);
        bi = bi < 0 ? 0 : (bi > 255 ? 255 : bi);
        float v = s_cs[bi];

        // pos = #{ j : cdf_tgt[j] <= v }  (branchless binary search)
        int pos = 0;
        #pragma unroll
        for (int s = 128; s >= 1; s >>= 1) {
            int np = pos + s;
            if (np <= NBINS && s_ct[np - 1] <= v) pos = np;
        }
        int idx = pos < 1 ? 1 : (pos > 255 ? 255 : pos);

        float x0 = s_ct[idx - 1];
        float x1 = s_ct[idx];
        float t  = (v - x0) / (x1 - x0 + 1e-6f);
        float y  = ((float)(idx - 1) + t) * (1.0f / 255.0f);
        out[base + i] = fminf(fmaxf(y, 0.f), 1.f);
    }
}

// ---------------------------------------------------------------------------
extern "C" void kernel_launch(void* const* d_in, const int* in_sizes, int n_in,
                              void* d_out, int out_size, void* d_ws, size_t ws_size,
                              hipStream_t stream) {
    (void)in_sizes; (void)n_in; (void)out_size; (void)ws_size;
    const float* src = (const float*)d_in[0];
    const float* tgt = (const float*)d_in[1];
    float* out = (float*)d_out;

    float* ws      = (float*)d_ws;
    float* partial = ws;                              // 12*CHUNKS*NBINS floats
    float* cdf_src = ws + 12 * CHUNKS * NBINS;        // 6*NBINS
    float* cdf_tgt = cdf_src + 6 * NBINS;             // 6*NBINS

    hist_wmma_kernel<<<12 * CHUNKS, HIST_THREADS, 0, stream>>>(src, tgt, partial);
    cdf_kernel<<<12, NBINS, 0, stream>>>(partial, cdf_src, cdf_tgt);
    map_kernel<<<6 * CHUNKS, NBINS, 0, stream>>>(src, cdf_src, cdf_tgt, out);
}
